// CompartmentAwareNormalization_60447369724330
// MI455X (gfx1250) — compile-verified, matrix operand order checked by reading, not run on hardware
//
#include <hip/hip_runtime.h>
#include <hip/hip_bf16.h>

// ---------------------------------------------------------------------------
// CompartmentAwareNormalization for MI455X (gfx1250, wave32, WMMA)
//   B=4, S=8192, D=1024, NC=5
//   Phase 1: per-token LayerNorm (+ per-compartment affine/scale) -> y
//            y stored fp32 in d_out (residual term) and bf16 in d_ws (GEMM A)
//   Phase 2: W fp32 -> bf16 in d_ws (GEMM B; 2MB, L2-resident)
//   Phase 3: out = y + y @ W^T + b via v_wmma_f32_16x16x32_bf16
// ---------------------------------------------------------------------------

#define D_MODEL 1024
#define NCOMP   5
#define LN_EPS  1e-5f

typedef __attribute__((ext_vector_type(16))) __bf16        v16bf;
typedef __attribute__((ext_vector_type(8)))  float         v8f;
typedef __attribute__((ext_vector_type(4)))  float         v4f;
typedef __attribute__((ext_vector_type(4)))  unsigned int  v4u;
typedef __attribute__((ext_vector_type(8)))  unsigned int  v8u;
typedef __attribute__((ext_vector_type(4)))  unsigned short v4us;

__device__ __forceinline__ unsigned short f32_to_bf16_rne(float f) {
    unsigned int u = __float_as_uint(f);
    unsigned int r = u + 0x7FFFu + ((u >> 16) & 1u);   // round-to-nearest-even
    return (unsigned short)(r >> 16);
}

// ---------------------------------------------------------------------------
// Kernel 1: LayerNorm. One wave (32 lanes) per token; 8 x float4 per lane.
// ---------------------------------------------------------------------------
__global__ __launch_bounds__(256) void ln_kernel(
    const float* __restrict__ x, const int* __restrict__ ids,
    const float* __restrict__ gamma, const float* __restrict__ beta,
    const float* __restrict__ scale,
    float* __restrict__ y_f32, unsigned short* __restrict__ y_bf16, int ntok)
{
    const int lane = threadIdx.x & 31;
    const int wave = threadIdx.x >> 5;
    const int t = blockIdx.x * 8 + wave;
    if (t >= ntok) return;

    const v4f* xt = (const v4f*)(x + (size_t)t * D_MODEL);
    v4f v[8];
    float sum = 0.f, sq = 0.f;
#pragma unroll
    for (int c = 0; c < 8; ++c) {
        v[c] = xt[c * 32 + lane];
        sum += v[c].x + v[c].y + v[c].z + v[c].w;
        sq  += v[c].x * v[c].x + v[c].y * v[c].y + v[c].z * v[c].z + v[c].w * v[c].w;
    }
#pragma unroll
    for (int m = 16; m >= 1; m >>= 1) {
        sum += __shfl_xor(sum, m, 32);
        sq  += __shfl_xor(sq,  m, 32);
    }
    const float inv_d = 1.0f / (float)D_MODEL;
    const float mu   = sum * inv_d;
    const float var  = sq * inv_d - mu * mu;
    const float rstd = rsqrtf(var + LN_EPS);

    const int  cid   = ids[t];
    const bool valid = (cid < NCOMP);                    // reference guard
    const int  cidc  = min(max(cid, 0), NCOMP - 1);
    const v4f* g4 = (const v4f*)(gamma + (size_t)cidc * D_MODEL);
    const v4f* b4 = (const v4f*)(beta  + (size_t)cidc * D_MODEL);
    const float s = scale[cidc];

    v4f*  yo = (v4f*)(y_f32 + (size_t)t * D_MODEL);
    v4us* yb = (v4us*)(y_bf16 + (size_t)t * D_MODEL);
#pragma unroll
    for (int c = 0; c < 8; ++c) {
        const int idx = c * 32 + lane;
        v4f yv;
        if (valid) {
            const v4f gv = g4[idx];
            const v4f bv = b4[idx];
            yv.x = ((v[c].x - mu) * rstd * gv.x + bv.x) * s;
            yv.y = ((v[c].y - mu) * rstd * gv.y + bv.y) * s;
            yv.z = ((v[c].z - mu) * rstd * gv.z + bv.z) * s;
            yv.w = ((v[c].w - mu) * rstd * gv.w + bv.w) * s;
        } else {
            yv = v[c];
        }
        yo[idx] = yv;
        v4us h;
        h.x = f32_to_bf16_rne(yv.x);
        h.y = f32_to_bf16_rne(yv.y);
        h.z = f32_to_bf16_rne(yv.z);
        h.w = f32_to_bf16_rne(yv.w);
        yb[idx] = h;
    }
}

// ---------------------------------------------------------------------------
// Kernel 2: W (fp32, [N,K] row-major) -> bf16
// ---------------------------------------------------------------------------
__global__ __launch_bounds__(256) void convw_kernel(
    const float* __restrict__ W, unsigned short* __restrict__ Wb, int n)
{
    const int i = (blockIdx.x * 256 + threadIdx.x) * 4;
    if (i < n) {
        const v4f w = *(const v4f*)(W + i);
        v4us h;
        h.x = f32_to_bf16_rne(w.x);
        h.y = f32_to_bf16_rne(w.y);
        h.z = f32_to_bf16_rne(w.z);
        h.w = f32_to_bf16_rne(w.w);
        *(v4us*)(Wb + i) = h;
    }
}

// ---------------------------------------------------------------------------
// Kernel 3: out[m,n] = y_f32[m,n] + bias[n] + sum_k A[m,k]*Bm[n,k]
//   A  = y bf16  [M, K] row-major
//   Bm = W bf16  [N, K] row-major  (WMMA B lane layout == W rows, k-contig)
//   Block: 8 waves -> 4 (M) x 2 (N); wave tile 32x64; block tile 128x128.
// ---------------------------------------------------------------------------
__device__ __forceinline__ v16bf load_frag(const unsigned short* p) {
    v8u c;
    c.lo = *(const v4u*)(p);        // k : [koff,   koff+8)
    c.hi = *(const v4u*)(p + 16);   // k : [koff+16, koff+24)
    return __builtin_bit_cast(v16bf, c);
}

__global__ __launch_bounds__(256) void gemm_kernel(
    const unsigned short* __restrict__ A,
    const unsigned short* __restrict__ Bm,
    const float* __restrict__ bias,
    float* __restrict__ out, int M)
{
    const int K = D_MODEL;            // also N == D_MODEL
    const int lane  = threadIdx.x & 31;
    const int wave  = threadIdx.x >> 5;
    const int waveM = wave & 3;
    const int waveN = wave >> 2;
    const int mBase = blockIdx.y * 128 + waveM * 32;
    const int nBase = blockIdx.x * 128 + waveN * 64;

    const int rowInFrag = lane & 15;
    const int koff      = (lane >> 4) * 8;   // 16-bit A/B operand lane layout

    const unsigned short* aPtr0 = A + (size_t)(mBase + rowInFrag) * K + koff;
    const unsigned short* aPtr1 = aPtr0 + (size_t)16 * K;
    const unsigned short* bPtr[4];
#pragma unroll
    for (int j = 0; j < 4; ++j)
        bPtr[j] = Bm + (size_t)(nBase + j * 16 + rowInFrag) * K + koff;

    const v8f vzero = {0.f, 0.f, 0.f, 0.f, 0.f, 0.f, 0.f, 0.f};
    v8f acc[2][4];
#pragma unroll
    for (int i = 0; i < 2; ++i)
#pragma unroll
        for (int j = 0; j < 4; ++j) acc[i][j] = vzero;

#pragma unroll 2
    for (int k = 0; k < K; k += 32) {
        // stream-ahead hint for the A rows (W stays L2-resident)
        __builtin_prefetch(aPtr0 + k + 256, 0, 0);
        __builtin_prefetch(aPtr1 + k + 256, 0, 0);

        v16bf a0 = load_frag(aPtr0 + k);
        v16bf a1 = load_frag(aPtr1 + k);
        v16bf b0 = load_frag(bPtr[0] + k);
        v16bf b1 = load_frag(bPtr[1] + k);
        v16bf b2 = load_frag(bPtr[2] + k);
        v16bf b3 = load_frag(bPtr[3] + k);

        acc[0][0] = __builtin_amdgcn_wmma_f32_16x16x32_bf16(false, a0, false, b0, (short)0, acc[0][0], false, false);
        acc[0][1] = __builtin_amdgcn_wmma_f32_16x16x32_bf16(false, a0, false, b1, (short)0, acc[0][1], false, false);
        acc[0][2] = __builtin_amdgcn_wmma_f32_16x16x32_bf16(false, a0, false, b2, (short)0, acc[0][2], false, false);
        acc[0][3] = __builtin_amdgcn_wmma_f32_16x16x32_bf16(false, a0, false, b3, (short)0, acc[0][3], false, false);
        acc[1][0] = __builtin_amdgcn_wmma_f32_16x16x32_bf16(false, a1, false, b0, (short)0, acc[1][0], false, false);
        acc[1][1] = __builtin_amdgcn_wmma_f32_16x16x32_bf16(false, a1, false, b1, (short)0, acc[1][1], false, false);
        acc[1][2] = __builtin_amdgcn_wmma_f32_16x16x32_bf16(false, a1, false, b2, (short)0, acc[1][2], false, false);
        acc[1][3] = __builtin_amdgcn_wmma_f32_16x16x32_bf16(false, a1, false, b3, (short)0, acc[1][3], false, false);
    }

    // Epilogue: C/D layout — lanes 0-15: M = r, N = lane; lanes 16-31: M = 8+r.
    const int colLane = lane & 15;
    const int rowSel  = (lane >> 4) * 8;
#pragma unroll
    for (int i = 0; i < 2; ++i) {
#pragma unroll
        for (int j = 0; j < 4; ++j) {
            const int col = nBase + j * 16 + colLane;
            const float bb = bias[col];
#pragma unroll
            for (int r = 0; r < 8; ++r) {
                const int row = mBase + i * 16 + rowSel + r;
                const size_t o = (size_t)row * D_MODEL + col;
                out[o] = out[o] + acc[i][j][r] + bb;   // out currently holds y (fp32)
            }
        }
    }
}

// ---------------------------------------------------------------------------
// Launch
// ---------------------------------------------------------------------------
extern "C" void kernel_launch(void* const* d_in, const int* in_sizes, int n_in,
                              void* d_out, int out_size, void* d_ws, size_t ws_size,
                              hipStream_t stream)
{
    const float* x     = (const float*)d_in[0];
    const int*   ids   = (const int*)  d_in[1];
    const float* gamma = (const float*)d_in[2];
    const float* beta  = (const float*)d_in[3];
    const float* scale = (const float*)d_in[4];
    const float* W     = (const float*)d_in[5];
    const float* bias  = (const float*)d_in[6];
    float*       out   = (float*)d_out;

    const int ntok = in_sizes[1];                 // B*S (32768)

    unsigned short* yb = (unsigned short*)d_ws;   // [ntok, D] bf16  (64 MB)
    unsigned short* Wb = yb + (size_t)ntok * D_MODEL;  // [D, D] bf16 (2 MB)

    // Phase 1: LayerNorm -> y (fp32 into d_out, bf16 into ws)
    ln_kernel<<<(ntok + 7) / 8, 256, 0, stream>>>(x, ids, gamma, beta, scale,
                                                  out, yb, ntok);
    // Phase 2: W -> bf16
    convw_kernel<<<(D_MODEL * D_MODEL / 4 + 255) / 256, 256, 0, stream>>>(
        W, Wb, D_MODEL * D_MODEL);
    // Phase 3: WMMA GEMM + residual + bias
    dim3 grid(D_MODEL / 128, ntok / 128);
    gemm_kernel<<<grid, 256, 0, stream>>>(yb, Wb, bias, out, ntok);
}
